// Inflow_70806830841968
// MI455X (gfx1250) — matrix operand order repaired and checked by reading, not compiled, not run on hardware
//
#include <hip/hip_runtime.h>

// Problem constants (match reference; launch code derives N/E from in_sizes)
#define H_    2
#define FIN_  128
#define D_    64
#define DIAL_ 256

typedef __attribute__((ext_vector_type(16))) __bf16         v16bf;
typedef __attribute__((ext_vector_type(8)))  unsigned int   v8u;
typedef __attribute__((ext_vector_type(8)))  float          v8f;

// ---------- helpers ----------
// One-time weight conversion: round-to-nearest-even
__device__ __forceinline__ unsigned short f2bf_rne(float x) {
  unsigned u = __builtin_bit_cast(unsigned, x);
  u += 0x7FFFu + ((u >> 16) & 1u);
  return (unsigned short)(u >> 16);
}

// Hot-path conversion: pack two f32 into two bf16 (truncate) with ONE v_perm_b32.
__device__ __forceinline__ unsigned pack2bf(float f0, float f1) {
  return __builtin_amdgcn_perm(__builtin_bit_cast(unsigned, f1),
                               __builtin_bit_cast(unsigned, f0), 0x07060302u);
}

__device__ __forceinline__ v8f wmma_bf16(v16bf a, v16bf b, v8f c) {
  return __builtin_amdgcn_wmma_f32_16x16x32_bf16(false, a, false, b, (short)0, c,
                                                 false, false);
}

// Load a 16x32 A tile (rows = lanes per ISA A layout) from row-major f32,
// converting to bf16 with a per-lane (per-row) scale (scale==1.f folds away).
__device__ __forceinline__ v16bf load_a_f32(const float* row0, int ld, int k0,
                                            int lane, float scale) {
  const int half = lane >> 4, m = lane & 15;
  const float* r = row0 + (size_t)m * ld + k0 + (half << 3);
  float4 x0 = *(const float4*)(r);
  float4 x1 = *(const float4*)(r + 4);
  float4 x2 = *(const float4*)(r + 16);
  float4 x3 = *(const float4*)(r + 20);
  v8u t;
  t[0] = pack2bf(x0.x * scale, x0.y * scale);
  t[1] = pack2bf(x0.z * scale, x0.w * scale);
  t[2] = pack2bf(x1.x * scale, x1.y * scale);
  t[3] = pack2bf(x1.z * scale, x1.w * scale);
  t[4] = pack2bf(x2.x * scale, x2.y * scale);
  t[5] = pack2bf(x2.z * scale, x2.w * scale);
  t[6] = pack2bf(x3.x * scale, x3.y * scale);
  t[7] = pack2bf(x3.z * scale, x3.w * scale);
  return __builtin_bit_cast(v16bf, t);
}

// Load a 32x16 B tile from a pre-converted bf16 (row-major KxN) weight buffer.
__device__ __forceinline__ v16bf load_b_bf(const unsigned short* wgt, int ld,
                                           int k0, int n0, int lane) {
  typedef __attribute__((ext_vector_type(16))) unsigned short v16us;
  const int half = lane >> 4, nn = lane & 15;
  v16us t;
#pragma unroll
  for (int s = 0; s < 16; ++s) {
    const int k = (s < 8) ? (8 * half + s) : (16 + 8 * half + (s - 8));
    t[s] = wgt[(size_t)(k0 + k) * ld + n0 + nn];
  }
  return __builtin_bit_cast(v16bf, t);
}

__device__ __forceinline__ void atomicMaxFloat(float* addr, float val) {
  unsigned* u = (unsigned*)addr;
  unsigned old = *u;
  while (true) {
    float f = __builtin_bit_cast(float, old);
    if (f >= val) break;
    unsigned assumed = old;
    old = atomicCAS(u, assumed, __builtin_bit_cast(unsigned, val));
    if (old == assumed) break;
  }
}

// ---------- kernels ----------
__global__ void cvt_bf16_kernel(const float* __restrict__ src,
                                unsigned short* __restrict__ dst, int n) {
  int i = blockIdx.x * blockDim.x + threadIdx.x;
  if (i < n) dst[i] = f2bf_rne(src[i]);
}

__global__ void init_kernel(float* __restrict__ emax, float* __restrict__ denom,
                            float* __restrict__ agg, int nh, int nhd) {
  int i = blockIdx.x * blockDim.x + threadIdx.x;
  if (i < nh) { emax[i] = -3.0e38f; denom[i] = 0.f; }
  if (i < nhd) agg[i] = 0.f;
}

// q/k/m = f @ w for all (mat, head) combos in one block.
// grid = N/16, block = 128 (wave = d-tile); A fragments hoisted (loaded once).
__global__ void node_proj_kernel(const float* __restrict__ feat,
                                 const unsigned short* __restrict__ wnode_bf,
                                 float* __restrict__ mout, float* __restrict__ qout,
                                 float* __restrict__ kout) {
  const int tile = blockIdx.x;
  const int wave = threadIdx.x >> 5, lane = threadIdx.x & 31;
  const float* fb = feat + (size_t)tile * 16 * FIN_;
  v16bf afrag[4];
#pragma unroll
  for (int kc = 0; kc < 4; ++kc)
    afrag[kc] = load_a_f32(fb, FIN_, kc * 32, lane, 1.f);
  const int half = lane >> 4, n = lane & 15;
#pragma unroll
  for (int combo = 0; combo < 6; ++combo) {
    const int mat = combo >> 1, h = combo & 1;
    const unsigned short* w = wnode_bf + ((size_t)mat * H_ + h) * (FIN_ * D_);
    v8f c = {};
#pragma unroll
    for (int kc = 0; kc < 4; ++kc)
      c = wmma_bf16(afrag[kc], load_b_bf(w, D_, kc * 32, wave * 16, lane), c);
    float* out = (mat == 0) ? mout : (mat == 1) ? qout : kout;
#pragma unroll
    for (int v = 0; v < 8; ++v) {
      int node = tile * 16 + v + 8 * half;
      out[((size_t)node * H_ + h) * D_ + wave * 16 + n] = c[v];
    }
  }
}

// Fused: rm = rel @ w_m (WMMA -> LDS), e = leaky(<k[src]+rm, q[dst]>), atomic-max per (dst,h)
// grid = E/16, block = 64 (wave = head); A fragments hoisted out of the d-tile loop.
__global__ void edge_score_kernel(const float* __restrict__ rel,
                                  const unsigned short* __restrict__ wm_bf,
                                  const float* __restrict__ qb,
                                  const float* __restrict__ kb,
                                  const int* __restrict__ src,
                                  const int* __restrict__ dst,
                                  float* __restrict__ e_out,
                                  float* __restrict__ emax) {
  __shared__ float rmLDS[H_][16][D_];
  const int tile = blockIdx.x;
  const int h = threadIdx.x >> 5, lane = threadIdx.x & 31;
  const float* relbase = rel + (size_t)tile * 16 * FIN_;
  const unsigned short* wmh = wm_bf + (size_t)h * FIN_ * D_;
  v16bf afrag[4];
#pragma unroll
  for (int kc = 0; kc < 4; ++kc)
    afrag[kc] = load_a_f32(relbase, FIN_, kc * 32, lane, 1.f);
  const int half = lane >> 4, n = lane & 15;
#pragma unroll
  for (int dt = 0; dt < 4; ++dt) {
    v8f c = {};
#pragma unroll
    for (int kc = 0; kc < 4; ++kc)
      c = wmma_bf16(afrag[kc], load_b_bf(wmh, D_, kc * 32, dt * 16, lane), c);
#pragma unroll
    for (int v = 0; v < 8; ++v) rmLDS[h][v + 8 * half][dt * 16 + n] = c[v];
  }
  __syncthreads();

  const int eidx = lane & 15, part = lane >> 4;
  const int eg = tile * 16 + eidx;
  const int s = src[eg], dd = dst[eg];
  const float* kp = kb + ((size_t)s * H_ + h) * D_ + part * 32;
  const float* qp = qb + ((size_t)dd * H_ + h) * D_ + part * 32;
  const float* rp = &rmLDS[h][eidx][part * 32];
  float acc = 0.f;
#pragma unroll
  for (int d4 = 0; d4 < 8; ++d4) {
    float4 kk = *(const float4*)(kp + 4 * d4);
    float4 qq = *(const float4*)(qp + 4 * d4);
    float4 rr = *(const float4*)(rp + 4 * d4);
    acc += (kk.x + rr.x) * qq.x + (kk.y + rr.y) * qq.y +
           (kk.z + rr.z) * qq.z + (kk.w + rr.w) * qq.w;
  }
  acc += __shfl_xor(acc, 16);
  if (part == 0) {
    float e = (acc >= 0.f) ? acc : 0.01f * acc;   // LeakyReLU(0.01)
    e_out[(size_t)eg * H_ + h] = e;
    atomicMaxFloat(&emax[(size_t)dd * H_ + h], e);
  }
}

// ex = exp(e - emax[dst]); denom[dst] += ex   (in-place overwrite of e)
__global__ void edge_exp_kernel(const int* __restrict__ dst, float* __restrict__ e,
                                float* __restrict__ denom,
                                const float* __restrict__ emax, int total) {
  int i = blockIdx.x * blockDim.x + threadIdx.x;
  if (i >= total) return;
  int eg = i >> 1, h = i & 1;  // H_ == 2
  int dd = dst[eg];
  float ex = __expf(e[i] - emax[(size_t)dd * H_ + h]);
  e[i] = ex;
  atomicAdd(&denom[(size_t)dd * H_ + h], ex);
}

// Fused: recompute rm (WMMA -> LDS); agg[dst] += (m[src]+rm) * ex
// grid = E/16, block = 64 (wave = head); A fragments hoisted.
__global__ void edge_msg_kernel(const float* __restrict__ rel,
                                const unsigned short* __restrict__ wm_bf,
                                const float* __restrict__ mb,
                                const int* __restrict__ src,
                                const int* __restrict__ dst,
                                const float* __restrict__ ex,
                                float* __restrict__ agg) {
  __shared__ float rmLDS[H_][16][D_];
  const int tile = blockIdx.x;
  const int h = threadIdx.x >> 5, lane = threadIdx.x & 31;
  const float* relbase = rel + (size_t)tile * 16 * FIN_;
  const unsigned short* wmh = wm_bf + (size_t)h * FIN_ * D_;
  v16bf afrag[4];
#pragma unroll
  for (int kc = 0; kc < 4; ++kc)
    afrag[kc] = load_a_f32(relbase, FIN_, kc * 32, lane, 1.f);
  const int half = lane >> 4, n = lane & 15;
#pragma unroll
  for (int dt = 0; dt < 4; ++dt) {
    v8f c = {};
#pragma unroll
    for (int kc = 0; kc < 4; ++kc)
      c = wmma_bf16(afrag[kc], load_b_bf(wmh, D_, kc * 32, dt * 16, lane), c);
#pragma unroll
    for (int v = 0; v < 8; ++v) rmLDS[h][v + 8 * half][dt * 16 + n] = c[v];
  }
  __syncthreads();

  const int eidx = lane & 15, part = lane >> 4;
  const int eg = tile * 16 + eidx;
  const int s = src[eg], dd = dst[eg];
  const float w = ex[(size_t)eg * H_ + h];
  const float* mp = mb + ((size_t)s * H_ + h) * D_ + part * 32;
  float* ap = agg + ((size_t)dd * H_ + h) * D_ + part * 32;
  const float* rp = &rmLDS[h][eidx][part * 32];
#pragma unroll
  for (int d4 = 0; d4 < 8; ++d4) {
    float4 mm = *(const float4*)(mp + 4 * d4);
    float4 rr = *(const float4*)(rp + 4 * d4);
    atomicAdd(&ap[4 * d4 + 0], (mm.x + rr.x) * w);
    atomicAdd(&ap[4 * d4 + 1], (mm.y + rr.y) * w);
    atomicAdd(&ap[4 * d4 + 2], (mm.z + rr.z) * w);
    atomicAdd(&ap[4 * d4 + 3], (mm.w + rr.w) * w);
  }
}

// out = (agg/denom) @ w_h_entity + dial @ w_h_dialogue
// grid = N/16, block = 128 (wave = d-tile)
__global__ void final_kernel(const float* __restrict__ agg,
                             const float* __restrict__ denom,
                             const float* __restrict__ dial,
                             const unsigned short* __restrict__ whe_bf,
                             const unsigned short* __restrict__ whd_bf,
                             float* __restrict__ out) {
  const int tile = blockIdx.x;
  const int wave = threadIdx.x >> 5, lane = threadIdx.x & 31;
  const int node0 = tile * 16;
  const int myrow = lane & 15;
  v8f c = {};
  const float* ab = agg + (size_t)node0 * (H_ * D_);
#pragma unroll
  for (int kc = 0; kc < 4; ++kc) {
    int h = (kc * 32) >> 6;                 // chunks never straddle heads
    float dn = denom[(size_t)(node0 + myrow) * H_ + h];
    float sc = (dn > 0.f) ? (1.f / dn) : 0.f;
    v16bf a = load_a_f32(ab, H_ * D_, kc * 32, lane, sc);
    v16bf b = load_b_bf(whe_bf, D_, kc * 32, wave * 16, lane);
    c = wmma_bf16(a, b, c);
  }
  const float* db = dial + (size_t)node0 * DIAL_;
#pragma unroll
  for (int kc = 0; kc < 8; ++kc) {
    v16bf a = load_a_f32(db, DIAL_, kc * 32, lane, 1.f);
    v16bf b = load_b_bf(whd_bf, D_, kc * 32, wave * 16, lane);
    c = wmma_bf16(a, b, c);
  }
  const int half = lane >> 4, n = lane & 15;
#pragma unroll
  for (int v = 0; v < 8; ++v) {
    int node = node0 + v + 8 * half;
    out[(size_t)node * D_ + wave * 16 + n] = c[v];
  }
}

// ---------- launch ----------
extern "C" void kernel_launch(void* const* d_in, const int* in_sizes, int n_in,
                              void* d_out, int out_size, void* d_ws, size_t ws_size,
                              hipStream_t stream) {
  const float* entity = (const float*)d_in[0];   // (N,1,128)
  const float* rel    = (const float*)d_in[1];   // (E,1,128)
  const float* dial   = (const float*)d_in[2];   // (N,256)
  const float* w_m    = (const float*)d_in[3];   // (2,128,64)
  const float* w_q    = (const float*)d_in[4];
  const float* w_k    = (const float*)d_in[5];
  const float* w_he   = (const float*)d_in[6];   // (128,64)
  const float* w_hd   = (const float*)d_in[7];   // (256,64)
  const int*   src    = (const int*)d_in[8];
  const int*   dst    = (const int*)d_in[9];
  float* out = (float*)d_out;

  const int N = in_sizes[0] / FIN_;
  const int E = in_sizes[8];
  const int NH = N * H_, NHD = N * H_ * D_, EH = E * H_;
  const int HFD = H_ * FIN_ * D_;                 // 16384

  // workspace carve-up (256B aligned)
  char* wsb = (char*)d_ws;
  size_t off = 0;
  auto take = [&](size_t bytes) {
    size_t r = off;
    off += (bytes + 255) & ~(size_t)255;
    return r;
  };
  unsigned short* wnode_bf = (unsigned short*)(wsb + take((size_t)3 * HFD * 2)); // m,q,k
  unsigned short* whe_bf   = (unsigned short*)(wsb + take((size_t)FIN_ * D_ * 2));
  unsigned short* whd_bf   = (unsigned short*)(wsb + take((size_t)DIAL_ * D_ * 2));
  float* qbuf  = (float*)(wsb + take((size_t)NHD * 4));
  float* kbuf  = (float*)(wsb + take((size_t)NHD * 4));
  float* mbuf  = (float*)(wsb + take((size_t)NHD * 4));
  float* ebuf  = (float*)(wsb + take((size_t)EH * 4));
  float* emax  = (float*)(wsb + take((size_t)NH * 4));
  float* denom = (float*)(wsb + take((size_t)NH * 4));
  float* aggb  = (float*)(wsb + take((size_t)NHD * 4));
  (void)ws_size; (void)n_in; (void)out_size;

  // 0) weight conversion + buffer init
  cvt_bf16_kernel<<<(HFD + 255) / 256, 256, 0, stream>>>(w_m, wnode_bf, HFD);
  cvt_bf16_kernel<<<(HFD + 255) / 256, 256, 0, stream>>>(w_q, wnode_bf + HFD, HFD);
  cvt_bf16_kernel<<<(HFD + 255) / 256, 256, 0, stream>>>(w_k, wnode_bf + 2 * HFD, HFD);
  cvt_bf16_kernel<<<(FIN_ * D_ + 255) / 256, 256, 0, stream>>>(w_he, whe_bf, FIN_ * D_);
  cvt_bf16_kernel<<<(DIAL_ * D_ + 255) / 256, 256, 0, stream>>>(w_hd, whd_bf, DIAL_ * D_);
  init_kernel<<<(NHD + 255) / 256, 256, 0, stream>>>(emax, denom, aggb, NH, NHD);

  // 1) node projections (WMMA bf16), all 6 (mat, head) combos per block
  node_proj_kernel<<<N / 16, 128, 0, stream>>>(entity, wnode_bf, mbuf, qbuf, kbuf);
  // 2) fused rm + edge scores + segment max
  edge_score_kernel<<<E / 16, 64, 0, stream>>>(rel, wnode_bf /* w_m at offset 0 */,
                                               qbuf, kbuf, src, dst, ebuf, emax);
  // 3) softmax numerator + denominator
  edge_exp_kernel<<<(EH + 255) / 256, 256, 0, stream>>>(dst, ebuf, denom, emax, EH);
  // 4) fused rm + weighted message scatter-add
  edge_msg_kernel<<<E / 16, 64, 0, stream>>>(rel, wnode_bf, mbuf, src, dst, ebuf, aggb);
  // 5) normalize + output GEMMs (WMMA bf16)
  final_kernel<<<N / 16, 128, 0, stream>>>(aggb, denom, dial, whe_bf, whd_bf, out);
}
